// CosineSimilarityAttention_69303592288271
// MI455X (gfx1250) — matrix operand order repaired and checked by reading, not compiled, not run on hardware
//
#include <hip/hip_runtime.h>
#include <hip/hip_bf16.h>

// CosineSimilarityAttention for MI455X (gfx1250, wave32, WMMA).
// B=4, Sq=Sk=4096, D=128. d_out = out[B,Sq,D] ++ attn[B,Sq,Sk] (fp32).
// d_ws usage: qn f16 (4MB) | knT f16 (4MB) | Vh f16 (4MB) = 12MB.

typedef __attribute__((ext_vector_type(16))) _Float16 v16h;
typedef __attribute__((ext_vector_type(8)))  _Float16 v8h;   // 16 bytes
typedef __attribute__((ext_vector_type(4)))  _Float16 v4h;   // 8 bytes
typedef __attribute__((ext_vector_type(8)))  float    v8f;

#define CSA_B  4
#define CSA_SQ 4096
#define CSA_SK 4096
#define CSA_D  128

// ---------------------------------------------------------------------------
// Kernel 1a: L2-normalize Q rows -> qn (f16, row major [B*Sq][D])
// ---------------------------------------------------------------------------
__global__ void csa_norm_q(const float* __restrict__ Q, _Float16* __restrict__ qn,
                           int rows) {
    int wave = (blockIdx.x * blockDim.x + threadIdx.x) >> 5;
    int lane = threadIdx.x & 31;
    if (wave >= rows) return;
    const float* row = Q + (size_t)wave * CSA_D;
    float4 v = *(const float4*)(row + lane * 4);
    float s = v.x * v.x + v.y * v.y + v.z * v.z + v.w * v.w;
    #pragma unroll
    for (int off = 16; off > 0; off >>= 1) s += __shfl_xor(s, off, 32);
    float sc = 1.0f / fmaxf(sqrtf(s), 1e-12f);
    v4h h;
    h[0] = (_Float16)(v.x * sc); h[1] = (_Float16)(v.y * sc);
    h[2] = (_Float16)(v.z * sc); h[3] = (_Float16)(v.w * sc);
    *(v4h*)(qn + (size_t)wave * CSA_D + lane * 4) = h;
}

// ---------------------------------------------------------------------------
// Kernel 1b: L2-normalize K rows -> knT (f16, transposed [B][D][Sk])
// ---------------------------------------------------------------------------
__global__ void csa_norm_k_t(const float* __restrict__ K, _Float16* __restrict__ knT,
                             int rows) {
    int wave = (blockIdx.x * blockDim.x + threadIdx.x) >> 5;
    int lane = threadIdx.x & 31;
    if (wave >= rows) return;
    int b = wave / CSA_SK;
    int k = wave % CSA_SK;
    const float* row = K + (size_t)wave * CSA_D;
    float4 v = *(const float4*)(row + lane * 4);
    float s = v.x * v.x + v.y * v.y + v.z * v.z + v.w * v.w;
    #pragma unroll
    for (int off = 16; off > 0; off >>= 1) s += __shfl_xor(s, off, 32);
    float sc = 1.0f / fmaxf(sqrtf(s), 1e-12f);
    _Float16* base = knT + ((size_t)b * CSA_D) * CSA_SK + k;
    int d0 = lane * 4;
    base[(size_t)(d0 + 0) * CSA_SK] = (_Float16)(v.x * sc);
    base[(size_t)(d0 + 1) * CSA_SK] = (_Float16)(v.y * sc);
    base[(size_t)(d0 + 2) * CSA_SK] = (_Float16)(v.z * sc);
    base[(size_t)(d0 + 3) * CSA_SK] = (_Float16)(v.w * sc);
}

// ---------------------------------------------------------------------------
// Kernel 1c: V f32 -> f16 copy (row major; a lane's PV B-fragment elements are D apart)
// ---------------------------------------------------------------------------
__global__ void csa_cvt_v(const float* __restrict__ V, _Float16* __restrict__ Vh,
                          int n4) {
    int i = blockIdx.x * blockDim.x + threadIdx.x;
    if (i >= n4) return;
    float4 v = ((const float4*)V)[i];
    v4h h;
    h[0] = (_Float16)v.x; h[1] = (_Float16)v.y;
    h[2] = (_Float16)v.z; h[3] = (_Float16)v.w;
    ((v4h*)Vh)[i] = h;
}

// ---------------------------------------------------------------------------
// Kernel 2: masked scores S = (qn . knT)/temp -> attn buffer (pre-softmax).
// One wave per 16x64 strip (4 N-tiles): A fragment loaded once per K-step and
// reused by 4 WMMAs. D/32 = 4 K-steps, 16 WMMA per wave.
// A layout (16b A 16x32): lane -> M=lane%16, K runs {8h..8h+7, 16+8h..23+8h}.
// B layout (16b B 32x16): lane -> N=lane%16, K runs {8h..8h+7, 16+8h..23+8h}, stride Sk.
// C layout: lane = 16h+N, VGPR r -> M=r+8h.
// ---------------------------------------------------------------------------
__global__ void csa_scores(const _Float16* __restrict__ qn,
                           const _Float16* __restrict__ knT,
                           const int* __restrict__ mask,
                           const float* __restrict__ temp,
                           float* __restrict__ attn) {
    const int kt64    = CSA_SK >> 6;                 // 64 strips of 64 cols
    const int qtiles  = CSA_SQ >> 4;
    const int tilesPerB = qtiles * kt64;
    int wid  = (blockIdx.x * blockDim.x + threadIdx.x) >> 5;
    int lane = threadIdx.x & 31;
    if (wid >= CSA_B * tilesPerB) return;
    int b  = wid / tilesPerB;
    int r  = wid % tilesPerB;
    int q0 = (r / kt64) << 4;
    int k0 = (r % kt64) << 6;

    int m    = lane & 15;
    int half = lane >> 4;

    const _Float16* arow  = qn  + ((size_t)b * CSA_SQ + q0 + m) * CSA_D;
    const _Float16* bbase = knT + ((size_t)b * CSA_D) * CSA_SK + k0;

    v8f c[4] = {};
    #pragma unroll
    for (int t = 0; t < CSA_D / 32; ++t) {
        v16h a;
        v8h r0 = *(const v8h*)(arow + 32 * t + 8 * half);
        v8h r1 = *(const v8h*)(arow + 32 * t + 16 + 8 * half);
        #pragma unroll
        for (int j = 0; j < 8; ++j) { a[j] = r0[j]; a[8 + j] = r1[j]; }
        const _Float16* brow = bbase + (size_t)(32 * t + 8 * half) * CSA_SK + m;
        #pragma unroll
        for (int jt = 0; jt < 4; ++jt) {
            const _Float16* bcol = brow + 16 * jt;
            v16h bf;
            #pragma unroll
            for (int j = 0; j < 8; ++j) { bf[j] = bcol[(size_t)j * CSA_SK]; bf[8 + j] = bcol[(size_t)(16 + j) * CSA_SK]; }
            c[jt] = __builtin_amdgcn_wmma_f32_16x16x32_f16(false, a, false, bf,
                                                           (short)0, c[jt],
                                                           false, false);
        }
    }

    float invT = 1.0f / temp[0];
    int n = lane & 15;
    #pragma unroll
    for (int jt = 0; jt < 4; ++jt) {
        #pragma unroll
        for (int rr = 0; rr < 8; ++rr) {
            int mm = rr + 8 * half;
            size_t idx = ((size_t)b * CSA_SQ + q0 + mm) * CSA_SK + k0 + 16 * jt + n;
            float s = c[jt][rr] * invT;
            // mask is 256MB of single-use data: keep it out of L2
            int mk = __builtin_nontemporal_load(mask + idx);
            if (mk == 0) s = -1e9f;
            attn[idx] = s;
        }
    }
}

// ---------------------------------------------------------------------------
// Kernel 3: row softmax over Sk=4096, in place. One 256-thread block per row,
// 16 fp32/thread in registers, wave shuffle + LDS for max/sum reductions.
// ---------------------------------------------------------------------------
__global__ void csa_softmax(float* __restrict__ attn) {
    __shared__ float red[8];
    const int tid = threadIdx.x;
    float* p = attn + (size_t)blockIdx.x * CSA_SK;

    float4 v[4];
    float mx = -3.0e38f;
    #pragma unroll
    for (int i = 0; i < 4; ++i) {
        v[i] = ((const float4*)p)[tid + 256 * i];
        mx = fmaxf(mx, fmaxf(fmaxf(v[i].x, v[i].y), fmaxf(v[i].z, v[i].w)));
    }
    #pragma unroll
    for (int off = 16; off > 0; off >>= 1) mx = fmaxf(mx, __shfl_xor(mx, off, 32));
    if ((tid & 31) == 0) red[tid >> 5] = mx;
    __syncthreads();
    if (tid == 0) {
        float m = red[0];
        #pragma unroll
        for (int i = 1; i < 8; ++i) m = fmaxf(m, red[i]);
        red[0] = m;
    }
    __syncthreads();
    mx = red[0];
    __syncthreads();   // red[0] consumed by all before reuse

    float sum = 0.f;
    #pragma unroll
    for (int i = 0; i < 4; ++i) {
        v[i].x = __expf(v[i].x - mx); sum += v[i].x;
        v[i].y = __expf(v[i].y - mx); sum += v[i].y;
        v[i].z = __expf(v[i].z - mx); sum += v[i].z;
        v[i].w = __expf(v[i].w - mx); sum += v[i].w;
    }
    #pragma unroll
    for (int off = 16; off > 0; off >>= 1) sum += __shfl_xor(sum, off, 32);
    if ((tid & 31) == 0) red[tid >> 5] = sum;
    __syncthreads();
    if (tid == 0) {
        float s = red[0];
        #pragma unroll
        for (int i = 1; i < 8; ++i) s += red[i];
        red[0] = s;
    }
    __syncthreads();
    float inv = 1.0f / red[0];

    #pragma unroll
    for (int i = 0; i < 4; ++i) {
        v[i].x *= inv; v[i].y *= inv; v[i].z *= inv; v[i].w *= inv;
        ((float4*)p)[tid + 256 * i] = v[i];
    }
}

// ---------------------------------------------------------------------------
// Kernel 4: out = attn @ V. One wave per 16x128 out strip (full D = 8 N-tiles)
// so attn is streamed exactly ONCE per q-tile. Sk/32 = 128 K-steps; per step
// one A fragment (fp32 attn -> f16 cvt) feeds 8 WMMAs; each lane reads column
// 16*jt+lane%16 of 16 Vh rows for the 8 B fragments (L2-resident, 4MB).
// ---------------------------------------------------------------------------
__global__ void csa_av(const float* __restrict__ attn,
                       const _Float16* __restrict__ Vh,
                       float* __restrict__ out) {
    const int qtiles = CSA_SQ >> 4;     // 256 strips per batch
    int wid  = (blockIdx.x * blockDim.x + threadIdx.x) >> 5;
    int lane = threadIdx.x & 31;
    if (wid >= CSA_B * qtiles) return;
    int b  = wid / qtiles;
    int q0 = (wid % qtiles) << 4;

    int m    = lane & 15;
    int half = lane >> 4;

    const float*    arow  = attn + ((size_t)b * CSA_SQ + q0 + m) * CSA_SK;
    const _Float16* vbase = Vh   + (size_t)b * CSA_SK * CSA_D;

    v8f c[8] = {};
    for (int t = 0; t < CSA_SK / 32; ++t) {
        const float* a0p = arow + 32 * t + 8 * half;
        const float* a1p = a0p + 16;
        float4 f0 = *(const float4*)(a0p);
        float4 f1 = *(const float4*)(a0p + 4);
        float4 f2 = *(const float4*)(a1p);
        float4 f3 = *(const float4*)(a1p + 4);
        v16h a;
        a[0]  = (_Float16)f0.x; a[1]  = (_Float16)f0.y;
        a[2]  = (_Float16)f0.z; a[3]  = (_Float16)f0.w;
        a[4]  = (_Float16)f1.x; a[5]  = (_Float16)f1.y;
        a[6]  = (_Float16)f1.z; a[7]  = (_Float16)f1.w;
        a[8]  = (_Float16)f2.x; a[9]  = (_Float16)f2.y;
        a[10] = (_Float16)f2.z; a[11] = (_Float16)f2.w;
        a[12] = (_Float16)f3.x; a[13] = (_Float16)f3.y;
        a[14] = (_Float16)f3.z; a[15] = (_Float16)f3.w;

        const _Float16* brow = vbase + (size_t)(32 * t + 8 * half) * CSA_D + m;
        #pragma unroll
        for (int jt = 0; jt < 8; ++jt) {
            const _Float16* bcol = brow + 16 * jt;
            v16h bf;
            #pragma unroll
            for (int j = 0; j < 8; ++j) { bf[j] = bcol[(size_t)j * CSA_D]; bf[8 + j] = bcol[(size_t)(16 + j) * CSA_D]; }
            c[jt] = __builtin_amdgcn_wmma_f32_16x16x32_f16(false, a, false, bf,
                                                           (short)0, c[jt],
                                                           false, false);
        }
    }

    int n = lane & 15;
    #pragma unroll
    for (int jt = 0; jt < 8; ++jt) {
        #pragma unroll
        for (int rr = 0; rr < 8; ++rr) {
            int mm = rr + 8 * half;
            out[((size_t)b * CSA_SQ + q0 + mm) * CSA_D + 16 * jt + n] = c[jt][rr];
        }
    }
}

// ---------------------------------------------------------------------------
extern "C" void kernel_launch(void* const* d_in, const int* in_sizes, int n_in,
                              void* d_out, int out_size, void* d_ws, size_t ws_size,
                              hipStream_t stream) {
    (void)in_sizes; (void)n_in; (void)out_size; (void)ws_size;

    const float* Q    = (const float*)d_in[0];
    const float* K    = (const float*)d_in[1];
    const float* V    = (const float*)d_in[2];
    const int*   mask = (const int*)d_in[3];
    const float* temp = (const float*)d_in[4];

    float* out  = (float*)d_out;                                  // [B,Sq,D]
    float* attn = (float*)d_out + (size_t)CSA_B * CSA_SQ * CSA_D; // [B,Sq,Sk]

    _Float16* qn  = (_Float16*)d_ws;                               // [B*Sq][D]
    _Float16* knT = qn  + (size_t)CSA_B * CSA_SQ * CSA_D;          // [B][D][Sk]
    _Float16* Vh  = knT + (size_t)CSA_B * CSA_D * CSA_SK;          // [B*Sk][D]

    const int qrows = CSA_B * CSA_SQ;   // 16384
    const int krows = CSA_B * CSA_SK;   // 16384

    // 1) normalize / convert (each wave = one row; 8 waves per block)
    csa_norm_q  <<<(qrows * 32 + 255) / 256, 256, 0, stream>>>(Q, qn, qrows);
    csa_norm_k_t<<<(krows * 32 + 255) / 256, 256, 0, stream>>>(K, knT, krows);
    {
        int n4 = CSA_B * CSA_SK * CSA_D / 4;  // 524288 float4
        csa_cvt_v<<<(n4 + 255) / 256, 256, 0, stream>>>(V, Vh, n4);
    }

    // 2) masked scores via WMMA: 65536 waves (16x64 strip each), 8 waves/block
    {
        int strips = CSA_B * (CSA_SQ >> 4) * (CSA_SK >> 6);
        csa_scores<<<(strips + 7) / 8, 256, 0, stream>>>(qn, knT, mask, temp, attn);
    }

    // 3) row softmax in place: one block per row
    csa_softmax<<<CSA_B * CSA_SQ, 256, 0, stream>>>(attn);

    // 4) out = attn @ V via WMMA: 1024 waves (16x128 strip each), 8 waves/block
    {
        int strips = CSA_B * (CSA_SQ >> 4);
        csa_av<<<(strips + 7) / 8, 256, 0, stream>>>(attn, Vh, out);
    }
}
